// MoELayer_2654289789355
// MI455X (gfx1250) — compile-verified, matrix operand order checked by reading, not run on hardware
//
#include <hip/hip_runtime.h>
#include <hip/hip_bf16.h>
#include <math.h>

// Problem constants from the reference: B=4, T=2048, D=1024, F=2048, E=16
#define N_TOK 8192
#define DIM   1024
#define FDIM  2048
#define NEXP  16
#define M_TILE 32
#define XS_STRIDE (DIM + 8)    // padded bf16 row stride to dodge bank conflicts
#define HS_STRIDE (FDIM + 8)

typedef __attribute__((ext_vector_type(16))) __bf16 v16bf;
typedef __attribute__((ext_vector_type(8)))  float  v8f;

// ---------------------------------------------------------------------------
// CDNA5 16-bit A-matrix 16x32 fragment layout (ISA 7.12.2): element e (0..15),
// v = e/2: k = (v>=4 ? 16 : 0) + (lane/16)*8 + (v%4)*2 + (e%2); row/col = lane%16.
// Within a lane-half, e=0..7 cover k+0..7 contiguously and e=8..15 cover
// k+16..23 -> two 16B runs, so LDS A-frag loads merge into 2x ds_load_b128.
// ---------------------------------------------------------------------------
__device__ __forceinline__ v16bf frag_lds(const __bf16* base, int stride,
                                          int row0, int k0) {
  const int lane = threadIdx.x & 31;
  const __bf16* p = base + (row0 + (lane & 15)) * stride + k0 + ((lane >> 4) << 3);
  v16bf f;
#pragma unroll
  for (int e = 0; e < 16; ++e) {
    const int v = e >> 1;
    f[e] = p[((v & 4) << 2) + ((v & 3) << 1) + (e & 1)];
  }
  return f;
}

// B fragments come from pre-tiled bf16 weights: tile (kt,nt) stores 512 bf16
// as [lane][e] so each lane reads its 16 elements as one 32B contiguous load.
__device__ __forceinline__ v16bf frag_b_tiled(const __bf16* __restrict__ Wt,
                                              int numNT, int kt, int nt) {
  const int lane = threadIdx.x & 31;
  const __bf16* p = Wt + (((size_t)(kt * numNT + nt)) << 9) + (lane << 4);
  return *(const v16bf*)p;
}

__device__ __forceinline__ float silu_f(float v) {
  return v * __builtin_amdgcn_rcpf(1.0f + __expf(-v));   // v_rcp_f32, no IEEE div
}

// ---------------------------------------------------------------------------
// Weight pre-convert: f32 row-major [E][K][N] -> bf16 fragment-tiled
// [E][kt][nt][lane][e], kt=K/32 tiles, nt=N/16 tiles.
// ---------------------------------------------------------------------------
extern "C" __global__ void __launch_bounds__(256)
moe_cvtw_kernel(const float* __restrict__ W, __bf16* __restrict__ Wt,
                int K, int N, long total) {
  const long gid = (long)blockIdx.x * blockDim.x + threadIdx.x;
  if (gid >= total) return;
  const int slot = (int)(gid & 511);
  const long tile = gid >> 9;
  const int numNT = N >> 4;
  const int tilesPerExp = (K >> 5) * numNT;
  const int ex = (int)(tile / tilesPerExp);
  const int tl = (int)(tile - (long)ex * tilesPerExp);
  const int kt = tl / numNT, nt = tl - kt * numNT;
  const int lane = slot >> 4, e = slot & 15;
  const int v = e >> 1;
  const int k = (kt << 5) + ((v & 4) << 2) + ((lane >> 4) << 3)
              + ((v & 3) << 1) + (e & 1);
  const int n = (nt << 4) + (lane & 15);
  Wt[gid] = (__bf16)W[((size_t)ex * K + k) * N + n];
}

// ---------------------------------------------------------------------------
// Gate: logits = x @ Wg, top-2, softmax over the two logits, expert counts
// ---------------------------------------------------------------------------
extern "C" __global__ void __launch_bounds__(128)
moe_gate_kernel(const float* __restrict__ x, const float* __restrict__ Wg,
                int* __restrict__ counts, int* __restrict__ eidx,
                float* __restrict__ ew) {
  __shared__ float red[128 * NEXP];
  const int t = blockIdx.x;
  const int tid = threadIdx.x;
  float acc[NEXP];
#pragma unroll
  for (int e = 0; e < NEXP; ++e) acc[e] = 0.f;
  const float* xt = x + (size_t)t * DIM;
  for (int d = tid; d < DIM; d += 128) {
    const float xv = xt[d];
    const float* wg = Wg + (size_t)d * NEXP;
#pragma unroll
    for (int e = 0; e < NEXP; ++e) acc[e] += xv * wg[e];
  }
#pragma unroll
  for (int e = 0; e < NEXP; ++e) red[tid * NEXP + e] = acc[e];
  __syncthreads();
  for (int s = 64; s > 0; s >>= 1) {
    if (tid < s) {
#pragma unroll
      for (int e = 0; e < NEXP; ++e)
        red[tid * NEXP + e] += red[(tid + s) * NEXP + e];
    }
    __syncthreads();
  }
  if (tid == 0) {
    int i0 = 0; float s0 = red[0];
    for (int e = 1; e < NEXP; ++e) { float v = red[e]; if (v > s0) { s0 = v; i0 = e; } }
    int i1 = (i0 == 0) ? 1 : 0; float s1 = red[i1];
    for (int e = 0; e < NEXP; ++e) {
      if (e == i0 || e == i1) continue;
      float v = red[e]; if (v > s1) { s1 = v; i1 = e; }
    }
    const float w0 = 1.0f / (1.0f + __expf(s1 - s0));
    eidx[2 * t] = i0; eidx[2 * t + 1] = i1;
    ew[2 * t] = w0;   ew[2 * t + 1] = 1.0f - w0;
    atomicAdd(&counts[i0], 1);
    atomicAdd(&counts[i1], 1);
  }
}

extern "C" __global__ void moe_zero_counts(int* __restrict__ counts) {
  if (threadIdx.x < NEXP) counts[threadIdx.x] = 0;
}

extern "C" __global__ void moe_scan_kernel(const int* __restrict__ counts,
                                           int* __restrict__ offsets,
                                           int* __restrict__ cursor) {
  if (threadIdx.x == 0) {
    int acc = 0;
    for (int e = 0; e < NEXP; ++e) { offsets[e] = acc; cursor[e] = acc; acc += counts[e]; }
    offsets[NEXP] = acc;
  }
}

extern "C" __global__ void __launch_bounds__(256)
moe_scatter_kernel(const int* __restrict__ eidx, int* __restrict__ cursor,
                   int* __restrict__ tokIdx, int* __restrict__ pos) {
  const int t = blockIdx.x * blockDim.x + threadIdx.x;
  if (t >= N_TOK) return;
  const int p0 = atomicAdd(&cursor[eidx[2 * t]], 1);
  tokIdx[p0] = t; pos[2 * t] = p0;
  const int p1 = atomicAdd(&cursor[eidx[2 * t + 1]], 1);
  tokIdx[p1] = t; pos[2 * t + 1] = p1;
}

// ---------------------------------------------------------------------------
// FFN: per-expert, 32-token tiles; 8 waves x (2 M-tiles x 4 N-tiles) = 8 WMMA
// per k-step.  H (32x2048 bf16) stays in LDS (CDNA5: 320 KB/WGP).
// ---------------------------------------------------------------------------
extern "C" __global__ void __launch_bounds__(256)
moe_ffn_kernel(const float* __restrict__ x, const __bf16* __restrict__ W1t,
               const __bf16* __restrict__ W2t, const int* __restrict__ offsets,
               const int* __restrict__ tokIdx, float* __restrict__ outBuf) {
  extern __shared__ char smem[];
  __bf16* Xs = (__bf16*)smem;                       // [M_TILE][XS_STRIDE]
  __bf16* Hs = Xs + M_TILE * XS_STRIDE;             // [M_TILE][HS_STRIDE]
  int*  sTok = (int*)(Hs + M_TILE * HS_STRIDE);     // [M_TILE]

  const int e = blockIdx.y;
  const int start = offsets[e];
  const int end   = offsets[e + 1];
  const int row0  = start + blockIdx.x * M_TILE;
  if (row0 >= end) return;

  const int tid = threadIdx.x;
  if (tid < M_TILE) {
    const int slot = row0 + tid;
    sTok[tid] = (slot < end) ? tokIdx[slot] : tokIdx[start];  // dup row; store guarded
  }
  __syncthreads();

  for (int i = tid; i < M_TILE * DIM; i += 256) {   // stage X tile as bf16
    const int m = i >> 10;
    const int k = i & (DIM - 1);
    Xs[m * XS_STRIDE + k] = (__bf16)x[(size_t)sTok[m] * DIM + k];
  }
  __syncthreads();

  const __bf16* W1e = W1t + (size_t)e * DIM * FDIM;   // fragment-tiled
  const __bf16* W2e = W2t + (size_t)e * FDIM * DIM;
  const int wave = tid >> 5;
  const int lane = tid & 31;
  const int nc_l = lane & 15;
  const int m_of = (lane >> 4) << 3;

  // ---- GEMM1: [32,1024] x [1024,2048] + SiLU -> Hs ----
  {
    const int numNT = FDIM >> 4;                    // 128
    for (int np = 0; np < FDIM / 512; ++np) {       // 4 passes of 512 cols
      const int nt0 = np * 32 + wave * 4;
      v8f c0[4] = {}, c1[4] = {};
      for (int kt = 0; kt < DIM / 32; ++kt) {
        const int k0 = kt << 5;
        const v16bf a0 = frag_lds(Xs, XS_STRIDE, 0,  k0);
        const v16bf a1 = frag_lds(Xs, XS_STRIDE, 16, k0);
#pragma unroll
        for (int j = 0; j < 4; ++j) {
          const v16bf b = frag_b_tiled(W1e, numNT, kt, nt0 + j);
          c0[j] = __builtin_amdgcn_wmma_f32_16x16x32_bf16(false, a0, false, b, (short)0, c0[j], false, false);
          c1[j] = __builtin_amdgcn_wmma_f32_16x16x32_bf16(false, a1, false, b, (short)0, c1[j], false, false);
        }
      }
      const int n0 = nt0 << 4;
#pragma unroll
      for (int j = 0; j < 4; ++j) {
#pragma unroll
        for (int r = 0; r < 8; ++r) {
          const int mA = r + m_of, mB = 16 + r + m_of;
          Hs[mA * HS_STRIDE + n0 + j * 16 + nc_l] = (__bf16)silu_f(c0[j][r]);
          Hs[mB * HS_STRIDE + n0 + j * 16 + nc_l] = (__bf16)silu_f(c1[j][r]);
        }
      }
    }
  }
  __syncthreads();

  // ---- GEMM2: [32,2048] x [2048,1024] -> outBuf (per-assignment slot) ----
  {
    const int numNT = DIM >> 4;                     // 64
    for (int np = 0; np < DIM / 512; ++np) {        // 2 passes of 512 cols
      const int nt0 = np * 32 + wave * 4;
      v8f c0[4] = {}, c1[4] = {};
      for (int kt = 0; kt < FDIM / 32; ++kt) {
        const int k0 = kt << 5;
        const v16bf a0 = frag_lds(Hs, HS_STRIDE, 0,  k0);
        const v16bf a1 = frag_lds(Hs, HS_STRIDE, 16, k0);
#pragma unroll
        for (int j = 0; j < 4; ++j) {
          const v16bf b = frag_b_tiled(W2e, numNT, kt, nt0 + j);
          c0[j] = __builtin_amdgcn_wmma_f32_16x16x32_bf16(false, a0, false, b, (short)0, c0[j], false, false);
          c1[j] = __builtin_amdgcn_wmma_f32_16x16x32_bf16(false, a1, false, b, (short)0, c1[j], false, false);
        }
      }
      const int n0 = nt0 << 4;
#pragma unroll
      for (int r = 0; r < 8; ++r) {
        const int mA = r + m_of, mB = 16 + r + m_of;
        if (row0 + mA < end) {
          float* o = outBuf + (size_t)(row0 + mA) * DIM + n0 + nc_l;
#pragma unroll
          for (int j = 0; j < 4; ++j) o[j * 16] = c0[j][r];
        }
        if (row0 + mB < end) {
          float* o = outBuf + (size_t)(row0 + mB) * DIM + n0 + nc_l;
#pragma unroll
          for (int j = 0; j < 4; ++j) o[j * 16] = c1[j][r];
        }
      }
    }
  }
}

// ---------------------------------------------------------------------------
// Combine (deterministic): y[t] = w0*outBuf[p0] + w1*outBuf[p1]
// ---------------------------------------------------------------------------
extern "C" __global__ void __launch_bounds__(256)
moe_combine_kernel(const float* __restrict__ outBuf, const int* __restrict__ pos,
                   const float* __restrict__ ew, float* __restrict__ y) {
  const int t = blockIdx.x;
  const int p0 = pos[2 * t], p1 = pos[2 * t + 1];
  const float w0 = ew[2 * t], w1 = ew[2 * t + 1];
  const float* b0 = outBuf + (size_t)p0 * DIM;
  const float* b1 = outBuf + (size_t)p1 * DIM;
  float* yt = y + (size_t)t * DIM;
  for (int d = threadIdx.x; d < DIM; d += 256)
    yt[d] = w0 * b0[d] + w1 * b1[d];
}

// ---------------------------------------------------------------------------
extern "C" void kernel_launch(void* const* d_in, const int* in_sizes, int n_in,
                              void* d_out, int out_size, void* d_ws, size_t ws_size,
                              hipStream_t stream) {
  const float* x  = (const float*)d_in[0];
  const float* Wg = (const float*)d_in[1];
  const float* W1 = (const float*)d_in[2];
  const float* W2 = (const float*)d_in[3];
  float* y = (float*)d_out;

  // Workspace: outBuf (67.1 MB) + bf16 tiled weights (2 x 67.1 MB) + metadata
  char* ws = (char*)d_ws;
  float* outBuf = (float*)ws;
  size_t ofs = (size_t)2 * N_TOK * DIM * sizeof(float);
  __bf16* W1t = (__bf16*)(ws + ofs); ofs += (size_t)NEXP * DIM * FDIM * sizeof(__bf16);
  __bf16* W2t = (__bf16*)(ws + ofs); ofs += (size_t)NEXP * FDIM * DIM * sizeof(__bf16);
  int* counts  = (int*)(ws + ofs); ofs += 16 * sizeof(int);
  int* offsets = (int*)(ws + ofs); ofs += 32 * sizeof(int);
  int* cursor  = (int*)(ws + ofs); ofs += 16 * sizeof(int);
  int* tokIdx  = (int*)(ws + ofs); ofs += (size_t)2 * N_TOK * sizeof(int);
  int* eidx    = (int*)(ws + ofs); ofs += (size_t)2 * N_TOK * sizeof(int);
  int* pos     = (int*)(ws + ofs); ofs += (size_t)2 * N_TOK * sizeof(int);
  float* ew    = (float*)(ws + ofs); ofs += (size_t)2 * N_TOK * sizeof(float);

  const long w1Total = (long)NEXP * DIM * FDIM;
  const long w2Total = (long)NEXP * FDIM * DIM;
  hipLaunchKernelGGL(moe_cvtw_kernel, dim3((unsigned)((w1Total + 255) / 256)), dim3(256),
                     0, stream, W1, W1t, DIM, FDIM, w1Total);
  hipLaunchKernelGGL(moe_cvtw_kernel, dim3((unsigned)((w2Total + 255) / 256)), dim3(256),
                     0, stream, W2, W2t, FDIM, DIM, w2Total);

  hipLaunchKernelGGL(moe_zero_counts, dim3(1), dim3(32), 0, stream, counts);
  hipLaunchKernelGGL(moe_gate_kernel, dim3(N_TOK), dim3(128), 0, stream,
                     x, Wg, counts, eidx, ew);
  hipLaunchKernelGGL(moe_scan_kernel, dim3(1), dim3(32), 0, stream,
                     counts, offsets, cursor);
  hipLaunchKernelGGL(moe_scatter_kernel, dim3(N_TOK / 256), dim3(256), 0, stream,
                     eidx, cursor, tokIdx, pos);

  const size_t ldsBytes = (size_t)M_TILE * XS_STRIDE * sizeof(__bf16)
                        + (size_t)M_TILE * HS_STRIDE * sizeof(__bf16)
                        + M_TILE * sizeof(int);  // ~198 KB < 320 KB/WGP
  hipLaunchKernelGGL(moe_ffn_kernel, dim3(N_TOK / M_TILE, NEXP), dim3(256),
                     ldsBytes, stream, x, W1t, W2t, offsets, tokIdx, outBuf);

  hipLaunchKernelGGL(moe_combine_kernel, dim3(N_TOK), dim3(256), 0, stream,
                     outBuf, pos, ew, y);
}
